// OrthogonalSylvester_ConvVAE_63926293233714
// MI455X (gfx1250) — compile-verified, matrix operand order checked by reading, not run on hardware
//
#include <hip/hip_runtime.h>
#include <hip/hip_bf16.h>

typedef __attribute__((ext_vector_type(16))) _Float16 v16h;
typedef __attribute__((ext_vector_type(8)))  float    v8f;

static __device__ __forceinline__ float lrelu_f(float v){ return v >= 0.f ? v : 0.01f*v; }

// ------------------------------------------------------------------
// small utility kernels
// ------------------------------------------------------------------
__global__ void k_zero(float* p, int n){
  int i = blockIdx.x*blockDim.x + threadIdx.x;
  if (i < n) p[i] = 0.f;
}

// ------------------------------------------------------------------
// Pack an f32 matrix S[rowsValid][ld] into CDNA5 WMMA 16-bit fragment
// layout: P[tile][kc][lane][16] halves, tile covers rows tile*16..+15,
// kc covers K-chunk kc*32..+31.  Lane layout per ISA 7.12.2:
//   lane = (row&15) | (khalf<<4); element e -> k offset
//   kin(e) = (v<4 ? 2v : 16+2(v-4)) + (e&1) + 8*khalf,  v = e>>1
// Used for both A (lane -> M) and B (lane -> N, S = W[N][K] row-major).
// ------------------------------------------------------------------
__global__ void k_pack_frag(const float* __restrict__ S, _Float16* __restrict__ P,
                            int nrows, int ld, int Kvalid, int Tiles, int KC)
{
  long long i = (long long)blockIdx.x*blockDim.x + threadIdx.x;
  long long tot = (long long)Tiles*KC*512;
  if (i >= tot) return;
  int e = (int)(i & 15); long long t = i >> 4;
  int lane = (int)(t & 31); t >>= 5;
  int kc = (int)(t % KC); int tile = (int)(t / KC);
  int row   = tile*16 + (lane & 15);
  int khalf = lane >> 4;
  int v = e >> 1, lo = e & 1;
  int kk = kc*32 + ((v < 4) ? 2*v : 16 + 2*(v-4)) + lo + 8*khalf;
  float val = (row < nrows && kk < Kvalid) ? S[(long long)row*ld + kk] : 0.f;
  P[i] = (_Float16)val;
}

// ------------------------------------------------------------------
// WMMA GEMM on pre-packed fragments.
// C[M][Np] = A*B (+bias). One wave per 16x16 C tile.
// Inner loop: one v16h (2x b128) load per operand + one v_wmma.
// ------------------------------------------------------------------
__global__ void k_gemm_wmma(const _Float16* __restrict__ Ap,
                            const _Float16* __restrict__ Bp,
                            const float* __restrict__ bias, int biasN,
                            float* __restrict__ C,
                            int Mtiles, int Ntiles, int KC, int Np)
{
  int gtid = blockIdx.x*blockDim.x + threadIdx.x;
  int wave = gtid >> 5;              // wave-uniform
  int lane = threadIdx.x & 31;
  if (wave >= Mtiles*Ntiles) return; // uniform exit: EXEC all-ones in live waves
  int nt = wave % Ntiles, mt = wave / Ntiles;

  const v16h* Av = (const v16h*)Ap + (long long)mt*KC*32 + lane;
  const v16h* Bv = (const v16h*)Bp + (long long)nt*KC*32 + lane;

  v8f acc = {};
#pragma unroll 4
  for (int kc = 0; kc < KC; ++kc){
    v16h a = Av[(long long)kc*32];
    v16h b = Bv[(long long)kc*32];
    acc = __builtin_amdgcn_wmma_f32_16x16x32_f16(false, a, false, b,
                                                 (short)0, acc, false, false);
  }

  // 16x16 f32 D layout: lanes 0-15 -> M = r, lanes 16-31 -> M = 8+r; N = lane&15
  int n = lane & 15;
  int mbase = (lane >> 4)*8;
  float bs = (bias != nullptr && (nt*16 + n) < biasN) ? bias[nt*16 + n] : 0.f;
#pragma unroll
  for (int r = 0; r < 8; ++r)
    C[(long long)(mt*16 + mbase + r)*Np + nt*16 + n] = acc[r] + bs;
}

// ------------------------------------------------------------------
// direct conv / transposed conv (NCHW, OIHW / IOHW), stride 1
// ------------------------------------------------------------------
__global__ void k_conv2d(const float* __restrict__ x, const float* __restrict__ w,
                         const float* __restrict__ bias, float* __restrict__ y,
                         int Bn, int Ci, int Hi, int Wi,
                         int Co, int KH, int KW, int ph, int pw, int Ho, int Wo)
{
  long long idx = (long long)blockIdx.x*blockDim.x + threadIdx.x;
  long long total = (long long)Bn*Co*Ho*Wo;
  if (idx >= total) return;
  int wo = (int)(idx % Wo); long long t = idx / Wo;
  int ho = (int)(t % Ho); t /= Ho;
  int co = (int)(t % Co); int b = (int)(t / Co);
  float acc = bias[co];
  for (int ci = 0; ci < Ci; ++ci)
    for (int u = 0; u < KH; ++u){
      int hi = ho + u - ph;
      if (hi < 0 || hi >= Hi) continue;
      for (int v = 0; v < KW; ++v){
        int wi = wo + v - pw;
        if (wi < 0 || wi >= Wi) continue;
        acc += x[(((long long)b*Ci + ci)*Hi + hi)*Wi + wi] *
               w[((co*Ci + ci)*KH + u)*KW + v];
      }
    }
  y[idx] = acc;
}

// y[b,o,p,q] = sum_i,u,v x[b,i,p+ph-u,q+pw-v] * w[i][o][u][v]  (convT, stride 1)
__global__ void k_convT2d(const float* __restrict__ x, const float* __restrict__ w,
                          const float* __restrict__ bias, float* __restrict__ y,
                          int Bn, int Ci, int Hi, int Wi,
                          int Co, int KH, int KW, int ph, int pw, int Ho, int Wo)
{
  long long idx = (long long)blockIdx.x*blockDim.x + threadIdx.x;
  long long total = (long long)Bn*Co*Ho*Wo;
  if (idx >= total) return;
  int wo = (int)(idx % Wo); long long t = idx / Wo;
  int ho = (int)(t % Ho); t /= Ho;
  int co = (int)(t % Co); int b = (int)(t / Co);
  float acc = bias[co];
  for (int ci = 0; ci < Ci; ++ci)
    for (int u = 0; u < KH; ++u){
      int hi = ho + ph - u;
      if (hi < 0 || hi >= Hi) continue;
      for (int v = 0; v < KW; ++v){
        int wi = wo + pw - v;
        if (wi < 0 || wi >= Wi) continue;
        acc += x[(((long long)b*Ci + ci)*Hi + hi)*Wi + wi] *
               w[((ci*Co + co)*KH + u)*KW + v];
      }
    }
  y[idx] = acc;
}

// ------------------------------------------------------------------
// BatchNorm2d: stats (sum,sumsq per channel) + apply
// ------------------------------------------------------------------
__global__ void k_bnstats2d(const float* __restrict__ x, float* __restrict__ stats,
                            int Bn, int C, long long HW)
{
  int c = blockIdx.y;
  long long per = (long long)Bn*HW;
  float s = 0.f, s2 = 0.f;
  for (long long i = (long long)blockIdx.x*blockDim.x + threadIdx.x; i < per;
       i += (long long)gridDim.x*blockDim.x){
    long long bb = i / HW, sp = i % HW;
    float v = x[(bb*C + c)*HW + sp];
    s += v; s2 += v*v;
  }
  __shared__ float sh[256], sh2[256];
  sh[threadIdx.x] = s; sh2[threadIdx.x] = s2; __syncthreads();
  for (int st = 128; st > 0; st >>= 1){
    if (threadIdx.x < st){ sh[threadIdx.x] += sh[threadIdx.x+st]; sh2[threadIdx.x] += sh2[threadIdx.x+st]; }
    __syncthreads();
  }
  if (threadIdx.x == 0){ atomicAdd(&stats[c], sh[0]); atomicAdd(&stats[C+c], sh2[0]); }
}

__global__ void k_bnapply2d(float* __restrict__ x, const float* __restrict__ stats,
                            const float* __restrict__ g, const float* __restrict__ be,
                            int Bn, int C, long long HW, int relu)
{
  long long idx = (long long)blockIdx.x*blockDim.x + threadIdx.x;
  long long total = (long long)Bn*C*HW;
  if (idx >= total) return;
  int c = (int)((idx / HW) % C);
  float cnt = (float)((long long)Bn*HW);
  float mu  = stats[c] / cnt;
  float var = stats[C+c] / cnt - mu*mu;
  float v = g[c]*(x[idx] - mu)*rsqrtf(var + 1e-5f) + be[c];
  x[idx] = relu ? lrelu_f(v) : v;
}

// BatchNorm1d over batch: thread per feature (coalesced column walk), in-place
__global__ void k_bn1d(float* __restrict__ x, const float* __restrict__ g,
                       const float* __restrict__ be, int Bn, int N, int ld, int relu)
{
  int j = blockIdx.x*blockDim.x + threadIdx.x;
  if (j >= N) return;
  float s = 0.f, s2 = 0.f;
  for (int r = 0; r < Bn; ++r){ float v = x[(long long)r*ld + j]; s += v; s2 += v*v; }
  float mu  = s / (float)Bn;
  float var = s2 / (float)Bn - mu*mu;
  float inv = rsqrtf(var + 1e-5f) * g[j];
  float bb  = be[j];
  for (int r = 0; r < Bn; ++r){
    float v = (x[(long long)r*ld + j] - mu)*inv + bb;
    x[(long long)r*ld + j] = relu ? lrelu_f(v) : v;
  }
}

// ------------------------------------------------------------------
// small dense layers: Y[b][n] = X[b][:K] . W[n][:K] + bias[n]; act 1 = tanh
// ------------------------------------------------------------------
__global__ void k_linear(const float* __restrict__ X, int ldx,
                         const float* __restrict__ W, const float* __restrict__ bias,
                         float* __restrict__ Y, int Bn, int N, int K, int act)
{
  int idx = blockIdx.x*blockDim.x + threadIdx.x;
  if (idx >= Bn*N) return;
  int n = idx % N, b = idx / N;
  float acc = bias[n];
  const float* xr = X + (long long)b*ldx;
  const float* wr = W + (long long)n*K;
  for (int k = 0; k < K; ++k) acc += xr[k]*wr[k];
  Y[(long long)b*N + n] = (act == 1) ? tanhf(acc) : acc;
}

__global__ void k_rownorm(const float* __restrict__ q, float* __restrict__ amat,
                          int rows, int len)
{
  int r = blockIdx.x*blockDim.x + threadIdx.x;
  if (r >= rows) return;
  const float* src = q + (long long)r*len;
  float s = 0.f;
  for (int i = 0; i < len; ++i){ float v = src[i]; s += v*v; }
  float inv = rsqrtf(s);
  float* dst = amat + (long long)r*len;
  for (int i = 0; i < len; ++i) dst[i] = src[i]*inv;
}

// ------------------------------------------------------------------
// Bjorck orthogonalization: 1 block per (24x5) matrix, 100 LDS-resident steps
// ------------------------------------------------------------------
__global__ void k_bjorck(float* __restrict__ amat, int steps)
{
  __shared__ float A[120], A2[120], G[25];
  int t = threadIdx.x;
  float* gA = amat + (long long)blockIdx.x*120;
  if (t < 120) A[t] = gA[t];
  __syncthreads();
  for (int it = 0; it < steps; ++it){
    if (t < 25){
      int m = t/5, n = t%5;
      float g = 0.f;
      for (int z = 0; z < 24; ++z) g += A[z*5+m]*A[z*5+n];
      G[t] = (m == n ? 1.5f : 0.f) - 0.5f*g;     // tmp = 1.5I - 0.5 A^T A
    }
    __syncthreads();
    if (t < 120){
      int z = t/5, n = t%5;
      float acc = 0.f;
      for (int m = 0; m < 5; ++m) acc += A[z*5+m]*G[m*5+n];
      A2[t] = acc;
    }
    __syncthreads();
    if (t < 120) A[t] = A2[t];
    __syncthreads();
  }
  if (t < 120) gA[t] = A[t];
}

// ------------------------------------------------------------------
// Sylvester flows: one thread per batch element (B=256)
// ------------------------------------------------------------------
__global__ void k_flow(const float* __restrict__ zmu, const float* __restrict__ zlv,
                       const float* __restrict__ eps, const float* __restrict__ amat,
                       const float* __restrict__ fd,  const float* __restrict__ d1t,
                       const float* __restrict__ d2t, const float* __restrict__ bv,
                       float* __restrict__ z0o, float* __restrict__ zko,
                       float* __restrict__ ldjo, float* __restrict__ zkws)
{
  int b = blockIdx.x*blockDim.x + threadIdx.x;
  if (b >= 256) return;
  float z[24];
  for (int i = 0; i < 24; ++i){
    float v = zmu[b*24+i] + eps[b*24+i]*expf(0.5f*zlv[b*24+i]);
    z[i] = v; z0o[b*24+i] = v;
  }
  float ldj = 0.f;
  for (int k = 0; k < 4; ++k){
    const float* qk = amat + (long long)(b*4 + k)*120;   // q_ortho[k][b] : (24,5)
    float s[5] = {0,0,0,0,0};
    for (int zz = 0; zz < 24; ++zz){
      float zv = z[zz];
      for (int j = 0; j < 5; ++j) s[j] += zv*qk[zz*5+j];
    }
    float h[5];
    for (int m = 0; m < 5; ++m){
      float acc = bv[b*20 + m*4 + k];
      for (int j = 0; j < 5; ++j){
        float r2mj = (m < j)  ? fd[((long long)b*25 + j*5 + m)*4 + k]
                   : (m == j) ? d2t[b*20 + m*4 + k] : 0.f;
        acc += s[j]*r2mj;
      }
      h[m] = tanhf(acc);
    }
    float tv[5];
    for (int j = 0; j < 5; ++j){
      float acc = 0.f;
      for (int m = 0; m < 5; ++m){
        float r1jm = (j < m)  ? fd[((long long)b*25 + j*5 + m)*4 + k]
                   : (j == m) ? d1t[b*20 + j*4 + k] : 0.f;
        acc += r1jm*h[m];
      }
      tv[j] = acc;
    }
    for (int zz = 0; zz < 24; ++zz){
      float acc = 0.f;
      for (int j = 0; j < 5; ++j) acc += qk[zz*5+j]*tv[j];
      z[zz] += acc;
    }
    for (int m = 0; m < 5; ++m){
      float dj = 1.f + (1.f - h[m]*h[m])*d1t[b*20+m*4+k]*d2t[b*20+m*4+k];
      ldj += logf(fabsf(dj));
    }
  }
  for (int i = 0; i < 24; ++i){ zko[b*24+i] = z[i]; zkws[b*24+i] = z[i]; }
  ldjo[b] = ldj;
}

// ------------------------------------------------------------------
// host orchestration
// ------------------------------------------------------------------
extern "C" void kernel_launch(void* const* d_in, const int* in_sizes, int n_in,
                              void* d_out, int out_size, void* d_ws, size_t ws_size,
                              hipStream_t stream)
{
  (void)in_sizes; (void)n_in; (void)out_size; (void)ws_size;
  const int Bn = 256, Hh = 64, Zz = 24, MID = 100, W3 = 67;
  const int QOUT = 8*Hh*W3;                   // 34304
  const long long E1 = 256LL*32*64*63;        // conv1 out
  const long long E2 = 256LL*16*64*65;        // conv2 out
  const long long E3 = 256LL*8*64*67;         // conv3 out / dec0 (== B*QOUT)
  const int KC1 = QOUT/32;                    // 1072 K-chunks for GEMM1
  const int KC2 = 4;                          // 128/32 K-chunks for GEMM2

  const float* x   = (const float*)d_in[0];
  const float* eps = (const float*)d_in[1];
#define P(i) ((const float*)d_in[i])

  // workspace layout
  char* ws = (char*)d_ws; size_t off = 0;
  auto take = [&](size_t bytes)->char*{ char* p = ws + off; off += (bytes + 255) & ~(size_t)255; return p; };
  float* buf1  = (float*)take((size_t)E1*4);
  float* buf2  = (float*)take((size_t)E2*4);
  float* buf3  = (float*)take((size_t)E3*4);
  float* c100  = (float*)take(256*112*4);
  float* fd    = (float*)take(256*100*4);
  float* d1t   = (float*)take(256*20*4);
  float* d2t   = (float*)take(256*20*4);
  float* bv    = (float*)take(256*20*4);
  float* qbuf  = (float*)take(256*480*4);
  float* amat  = (float*)take(1024*120*4);
  float* zkbuf = (float*)take(256*24*4);
  float* d1v   = (float*)take(256*100*4);
  float* stats = (float*)take(64*4);
  _Float16* apack2 = (_Float16*)take(16*KC2*512*2);           // 32768 halfs
  // fragment staging overlaid on buf1 (dead between encoder conv2 and decoder convT2)
  _Float16* apack1 = (_Float16*)buf1;                         // 16*1072*512 = 8,781,824 halfs
  _Float16* bpack  = (_Float16*)((char*)buf1 + 17563648);     // up to 2144*4*512 = 4,390,912 halfs

  // output layout: x_dec | z_mu | z_lv | ldj | z0 | zk
  float* out   = (float*)d_out;
  float* o_xd  = out;
  float* o_zmu = out + 1048576;
  float* o_zlv = o_zmu + 6144;
  float* o_ldj = o_zlv + 6144;
  float* o_z0  = o_ldj + 256;
  float* o_zk  = o_z0  + 6144;

  auto nb = [](long long n)->unsigned{ return (unsigned)((n + 255)/256); };

  // ---------------- encoder ----------------
  k_conv2d<<<nb(E1),256,0,stream>>>(x, P(2), P(3), buf1, Bn,1,64,64, 32,3,4,1,1, 64,63);
  k_zero<<<1,64,0,stream>>>(stats, 64);
  k_bnstats2d<<<dim3(64,32),256,0,stream>>>(buf1, stats, Bn,32, 64LL*63);
  k_bnapply2d<<<nb(E1),256,0,stream>>>(buf1, stats, P(4), P(5), Bn,32, 64LL*63, 1);

  k_conv2d<<<nb(E2),256,0,stream>>>(buf1, P(6), P(7), buf2, Bn,32,64,63, 16,3,1,1,1, 64,65);
  k_zero<<<1,64,0,stream>>>(stats, 64);
  k_bnstats2d<<<dim3(64,16),256,0,stream>>>(buf2, stats, Bn,16, 64LL*65);
  k_bnapply2d<<<nb(E2),256,0,stream>>>(buf2, stats, P(8), P(9), Bn,16, 64LL*65, 1);

  k_conv2d<<<nb(E3),256,0,stream>>>(buf2, P(10), P(11), buf3, Bn,16,64,65, 8,3,1,1,1, 64,67);
  k_zero<<<1,64,0,stream>>>(stats, 64);
  k_bnstats2d<<<dim3(64,8),256,0,stream>>>(buf3, stats, Bn,8, 64LL*67);
  k_bnapply2d<<<nb(E3),256,0,stream>>>(buf3, stats, P(12), P(13), Bn,8, 64LL*67, 0);

  // ------- big GEMM 1: [256,34304] x [34304,100] via WMMA fragments -------
  k_pack_frag<<<nb(16LL*KC1*512),256,0,stream>>>(buf3,  apack1, 256, QOUT, QOUT, 16, KC1);
  k_pack_frag<<<nb(7LL*KC1*512),256,0,stream>>>(P(14), bpack,  100, QOUT, QOUT,  7, KC1);
  k_gemm_wmma<<<nb(16LL*7*32),256,0,stream>>>(apack1, bpack, P(15), 100, c100, 16, 7, KC1, 112);
  k_bn1d<<<nb(100),256,0,stream>>>(c100, P(16), P(17), Bn, 100, 112, 1);

  // ---------------- heads ----------------
  k_linear<<<nb(256*24),256,0,stream>>>(c100,112, P(18),P(19), o_zmu, Bn, 24, MID, 0);
  k_linear<<<nb(256*24),256,0,stream>>>(c100,112, P(20),P(21), o_zlv, Bn, 24, MID, 0);
  k_linear<<<nb(256*100),256,0,stream>>>(c100,112, P(22),P(23), fd,   Bn,100, MID, 0);
  k_linear<<<nb(256*20),256,0,stream>>>(c100,112, P(24),P(25), d1t,  Bn, 20, MID, 1);
  k_linear<<<nb(256*20),256,0,stream>>>(c100,112, P(26),P(27), d2t,  Bn, 20, MID, 1);
  k_linear<<<nb(256*20),256,0,stream>>>(c100,112, P(30),P(31), bv,   Bn, 20, MID, 0);
  k_linear<<<nb(256*480),256,0,stream>>>(c100,112, P(28),P(29), qbuf, Bn,480, MID, 0);

  // ---------------- orthogonalization + flows ----------------
  k_rownorm<<<nb(1024),256,0,stream>>>(qbuf, amat, 1024, 120);
  k_bjorck<<<1024,128,0,stream>>>(amat, 100);
  k_flow<<<1,256,0,stream>>>(o_zmu, o_zlv, eps, amat, fd, d1t, d2t, bv,
                             o_z0, o_zk, o_ldj, zkbuf);

  // ---------------- decoder ----------------
  k_linear<<<nb(256*100),256,0,stream>>>(zkbuf,Zz, P(32),P(33), d1v, Bn, 100, Zz, 0);
  k_bn1d<<<nb(100),256,0,stream>>>(d1v, P(34), P(35), Bn, 100, 100, 1);

  // ------- big GEMM 2: [256,100] x [100,34304] via WMMA fragments (K pad 128) -------
  k_pack_frag<<<nb(16LL*KC2*512),256,0,stream>>>(d1v,   apack2, 256,  100, 100,   16, KC2);
  k_pack_frag<<<nb(2144LL*KC2*512),256,0,stream>>>(P(36), bpack, QOUT, 100, 100, 2144, KC2);
  k_gemm_wmma<<<nb(16LL*2144*32),256,0,stream>>>(apack2, bpack, P(37), QOUT, buf3, 16, 2144, KC2, QOUT);

  k_bn1d<<<nb(QOUT),256,0,stream>>>(buf3, P(38), P(39), Bn, QOUT, QOUT, 0);
  k_zero<<<1,64,0,stream>>>(stats, 64);
  k_bnstats2d<<<dim3(64,8),256,0,stream>>>(buf3, stats, Bn,8, 64LL*67);
  k_bnapply2d<<<nb(E3),256,0,stream>>>(buf3, stats, P(40), P(41), Bn,8, 64LL*67, 0);

  k_convT2d<<<nb(E2),256,0,stream>>>(buf3, P(42), P(43), buf2, Bn,8,64,67, 16,3,1,1,1, 64,65);
  k_zero<<<1,64,0,stream>>>(stats, 64);
  k_bnstats2d<<<dim3(64,16),256,0,stream>>>(buf2, stats, Bn,16, 64LL*65);
  k_bnapply2d<<<nb(E2),256,0,stream>>>(buf2, stats, P(44), P(45), Bn,16, 64LL*65, 1);

  k_convT2d<<<nb(E1),256,0,stream>>>(buf2, P(46), P(47), buf1, Bn,16,64,65, 32,3,1,1,1, 64,63);
  k_zero<<<1,64,0,stream>>>(stats, 64);
  k_bnstats2d<<<dim3(64,32),256,0,stream>>>(buf1, stats, Bn,32, 64LL*63);
  k_bnapply2d<<<nb(E1),256,0,stream>>>(buf1, stats, P(48), P(49), Bn,32, 64LL*63, 1);

  k_convT2d<<<nb(256LL*64*64),256,0,stream>>>(buf1, P(50), P(51), o_xd, Bn,32,64,63, 1,3,4,1,1, 64,64);
#undef P
}
